// ConstrainedDiacriticsModel_2585570312768
// MI455X (gfx1250) — compile-verified, matrix operand order checked by reading, not run on hardware
//
#include <hip/hip_runtime.h>
#include <hip/hip_bf16.h>
#include <math.h>

// ---------------------------------------------------------------------------
// MI455X (gfx1250): all GEMMs via v_wmma_f32_16x16x32_f16 (wave32, f32 acc).
// Round-2 changes:
//  * K-outer loop order: each A fragment loaded once, shared by all column
//    tiles of the wave (8x fewer A loads in LSTM, 2x in GEMM, 8x/4x in attn).
//  * Whh (128KB f16) cached in LDS for the whole 100-step recurrence
//    (static LDS ~154KB, exercising CDNA5's 320KB/WGP LDS).
//  * lower register pressure -> fewer s_set_vgpr_msb / spill SALU.
// ---------------------------------------------------------------------------

typedef __attribute__((ext_vector_type(16))) _Float16 v16h;
typedef __attribute__((ext_vector_type(8)))  _Float16 v8h;
typedef __attribute__((ext_vector_type(8)))  float    v8f;

#define T_STEPS 100
#define NBATCH  1024     // B*S
#define HID     128
#define GATES   512
#define DMODEL  256
#define SEQ     128
#define DKEY    64

static __device__ inline float sigm(float x) { return 1.f / (1.f + __expf(-x)); }

// A-fragment (16x32, f16) from row-major matrix, leading dim ld.
// lanes 0-15 = rows 0-15 holding K {0..7,16..23}; lanes 16-31 same rows,
// K {8..15,24..31}.  (Same layout serves B = W^T when W is N x K row-major.)
static __device__ inline v16h load_frag(const _Float16* p, int ld) {
  int lane = threadIdx.x & 31;
  int m    = lane & 15;
  int kb   = (lane >> 4) << 3;   // 0 or 8
  const _Float16* row = p + (size_t)m * ld;
  v16h a;
#pragma unroll
  for (int i = 0; i < 8; ++i) a[i]     = row[kb + i];
#pragma unroll
  for (int i = 0; i < 8; ++i) a[i + 8] = row[kb + 16 + i];
  return a;
}
// B-fragment (32x16) where B is stored row-major K x N (the V matrix).
static __device__ inline v16h load_frag_kn(const _Float16* p, int ld) {
  int lane = threadIdx.x & 31;
  int n    = lane & 15;
  int kb   = (lane >> 4) << 3;
  v16h b;
#pragma unroll
  for (int i = 0; i < 8; ++i) b[i]     = p[(size_t)(kb + i) * ld + n];
#pragma unroll
  for (int i = 0; i < 8; ++i) b[i + 8] = p[(size_t)(kb + 16 + i) * ld + n];
  return b;
}

static __device__ inline v8f wmma16(v16h a, v16h b, v8f c) {
  return __builtin_amdgcn_wmma_f32_16x16x32_f16(false, a, false, b,
                                                (short)0, c, false, false);
}

// ---------------------------------------------------------------------------
__global__ void cvt_f16_kernel(const float* __restrict__ s,
                               _Float16* __restrict__ d, int n) {
  int i = blockIdx.x * 256 + threadIdx.x;
  if (i < n) d[i] = (_Float16)s[i];
}

// x0[(t*N + n)*128 + k] = emb[ctx[n*100 + t]][k]
__global__ void embed_kernel(const int* __restrict__ ctx,
                             const float* __restrict__ emb,
                             _Float16* __restrict__ x0) {
  int tn = blockIdx.x;                // t*NBATCH + n
  int t = tn / NBATCH, n = tn % NBATCH;
  int code = ctx[n * T_STEPS + t];
  x0[(size_t)tn * HID + threadIdx.x] = (_Float16)emb[code * HID + threadIdx.x];
}

// ---------------------------------------------------------------------------
// Persistent BiLSTM layer. grid = (NBATCH/32, 2 directions), block = 256.
// Wave w: row subtile mt=w&1, 32 gate columns cg=(w>>1)*32 replicated across
// the 4 gates -> sigmoid/tanh combine is register-local. Whh lives in LDS for
// the whole recurrence; c-state lives in VGPRs in WMMA C-fragment layout.
__global__ __launch_bounds__(256) void lstm_kernel(
    const _Float16* __restrict__ x, int in_dim,
    const _Float16* __restrict__ wih_f, const _Float16* __restrict__ whh_f,
    const float* __restrict__ bih_f, const float* __restrict__ bhh_f,
    const _Float16* __restrict__ wih_b, const _Float16* __restrict__ whh_b,
    const float* __restrict__ bih_b, const float* __restrict__ bhh_b,
    _Float16* __restrict__ out_h,      // (T,N,256) f16 or null
    float* __restrict__ center_out)    // (N,256) f32 or null (t == T/2 only)
{
  __shared__ _Float16 s_whh[GATES * HID];   // 128 KB: resident all 100 steps
  __shared__ _Float16 s_xt[32 * 256];       // 16 KB
  __shared__ _Float16 s_h [32 * HID];       //  8 KB
  __shared__ float    s_bias[GATES];        //  2 KB

  int dir = blockIdx.y;
  const _Float16* wih = dir ? wih_b : wih_f;
  const _Float16* whh = dir ? whh_b : whh_f;
  const float*    bih = dir ? bih_b : bih_f;
  const float*    bhh = dir ? bhh_b : bhh_f;
  int col0 = dir * HID;

  int tid = threadIdx.x, lane = tid & 31, wave = tid >> 5;
  int n0 = blockIdx.x * 32;
  int mt = wave & 1;          // 16-row subtile
  int cg = (wave >> 1) * 32;  // 32 gate columns owned by this wave
  int hi = lane >> 4, nn = lane & 15;

  {  // Whh -> LDS (16B vector copies), h := 0, bias := bih+bhh
    const v8h* wsrc = (const v8h*)whh;
    v8h* wdst = (v8h*)s_whh;
    for (int i = tid; i < (GATES * HID) / 8; i += 256) wdst[i] = wsrc[i];
  }
  for (int i = tid; i < 32 * HID; i += 256) s_h[i] = (_Float16)0.f;
  for (int i = tid; i < GATES;    i += 256) s_bias[i] = bih[i] + bhh[i];
  float cst[16];
#pragma unroll
  for (int i = 0; i < 16; ++i) cst[i] = 0.f;
  __syncthreads();

  for (int step = 0; step < T_STEPS; ++step) {
    int t = dir ? (T_STEPS - 1 - step) : step;
    // stage x_t tile (32 x in_dim)
    const v8h* src = (const v8h*)(x + ((size_t)t * NBATCH + n0) * in_dim);
    v8h* dst = (v8h*)s_xt;
    int nv = (32 * in_dim) / 8;
    for (int i = tid; i < nv; i += 256) dst[i] = src[i];
    // prefetch next timestep tile (global_prefetch_b8)
    if (step + 1 < T_STEPS) {
      int t2 = dir ? (T_STEPS - 2 - step) : step + 1;
      const char* pf = (const char*)(x + ((size_t)t2 * NBATCH + n0) * in_dim);
      if (tid * 64 < 32 * in_dim * 2) __builtin_prefetch(pf + tid * 64, 0, 1);
    }
    __syncthreads();

    // z tiles owned by this wave: q = gi*2+sb (gate gi, 16-col subtile sb)
    v8f acc[8] = {};
    const _Float16* arow_x = s_xt + mt * 16 * in_dim;
    for (int kk = 0; kk < in_dim; kk += 32) {          // x_t @ Wih^T
      v16h af = load_frag(arow_x + kk, in_dim);
#pragma unroll
      for (int q = 0; q < 8; ++q) {
        int ncol = (q >> 1) * HID + cg + (q & 1) * 16;
        v16h bf = load_frag(wih + (size_t)ncol * in_dim + kk, in_dim);
        acc[q] = wmma16(af, bf, acc[q]);
      }
    }
    const _Float16* arow_h = s_h + mt * 16 * HID;
#pragma unroll
    for (int kk = 0; kk < HID; kk += 32) {             // h @ Whh^T (LDS)
      v16h af = load_frag(arow_h + kk, HID);
#pragma unroll
      for (int q = 0; q < 8; ++q) {
        int ncol = (q >> 1) * HID + cg + (q & 1) * 16;
        v16h bf = load_frag(s_whh + ncol * HID + kk, HID);
        acc[q] = wmma16(af, bf, acc[q]);
      }
    }
    __syncthreads();   // all reads of s_h done before overwrite

#pragma unroll
    for (int sb = 0; sb < 2; ++sb) {
      int colbase = cg + sb * 16 + nn;
      float bi = s_bias[0 * HID + colbase];
      float bfo = s_bias[1 * HID + colbase];
      float bg = s_bias[2 * HID + colbase];
      float bo2 = s_bias[3 * HID + colbase];
#pragma unroll
      for (int r = 0; r < 8; ++r) {
        int row = mt * 16 + r + 8 * hi;
        float iz = acc[0 + sb][r] + bi;
        float fz = acc[2 + sb][r] + bfo;
        float gz = acc[4 + sb][r] + bg;
        float oz = acc[6 + sb][r] + bo2;
        float c = sigm(fz) * cst[sb * 8 + r] + sigm(iz) * tanhf(gz);
        cst[sb * 8 + r] = c;
        float h = sigm(oz) * tanhf(c);
        s_h[row * HID + colbase] = (_Float16)h;
        if (out_h)
          out_h[((size_t)t * NBATCH + n0 + row) * DMODEL + col0 + colbase] =
              (_Float16)h;
        if (center_out && t == T_STEPS / 2)
          center_out[(size_t)(n0 + row) * DMODEL + col0 + colbase] = h;
      }
    }
    __syncthreads();
  }
}

// ---------------------------------------------------------------------------
// Generic C = A(MxK,f16) * W(NxK,f16)^T + bias [+residual] [relu]
// grid = (M/64, N/64), block = 256. Wave w: rows (w>>1)*16, two 16-col tiles
// at (w&1)*32 -> both tiles share each A fragment.
__global__ __launch_bounds__(256) void gemm_kernel(
    const _Float16* __restrict__ A, int lda,
    const _Float16* __restrict__ W, int ldw,
    const float* __restrict__ bias,
    const float* __restrict__ residual, int ldr,
    float* __restrict__ Cf, _Float16* __restrict__ Ch, int ldc,
    int K, int relu)
{
  int lane = threadIdx.x & 31, wave = threadIdx.x >> 5;
  int hi = lane >> 4, nn = lane & 15;
  int m0 = blockIdx.x * 64 + (wave >> 1) * 16;
  int nbase = blockIdx.y * 64 + (wave & 1) * 32;

  v8f acc[2] = {};
  for (int kk = 0; kk < K; kk += 32) {
    v16h af = load_frag(A + (size_t)m0 * lda + kk, lda);
#pragma unroll
    for (int j = 0; j < 2; ++j) {
      v16h bf = load_frag(W + (size_t)(nbase + j * 16) * ldw + kk, ldw);
      acc[j] = wmma16(af, bf, acc[j]);
    }
  }
#pragma unroll
  for (int j = 0; j < 2; ++j) {
    int n = nbase + j * 16 + nn;
    float bv = bias ? bias[n] : 0.f;
#pragma unroll
    for (int r = 0; r < 8; ++r) {
      int m = m0 + r + 8 * hi;
      float v = acc[j][r] + bv;
      if (residual) v += residual[(size_t)m * ldr + n];
      if (relu) v = v > 0.f ? v : 0.f;
      if (Cf) Cf[(size_t)m * ldc + n] = v;
      if (Ch) Ch[(size_t)m * ldc + n] = (_Float16)v;
    }
  }
}

// ---------------------------------------------------------------------------
// One block per (batch b, head h): S=128, dk=64. scores->softmax->ctx.
__global__ __launch_bounds__(256) void attn_kernel(
    const _Float16* __restrict__ Q, const _Float16* __restrict__ Km,
    const _Float16* __restrict__ V, _Float16* __restrict__ ctx_h)
{
  __shared__ _Float16 s_p[SEQ * SEQ];   // 32KB: scores then probs (f16)
  int b = blockIdx.x >> 2, h = blockIdx.x & 3;
  int lane = threadIdx.x & 31, wave = threadIdx.x >> 5;
  int hi = lane >> 4, nn = lane & 15;
  const _Float16* Qb = Q + (size_t)(b * SEQ) * DMODEL + h * DKEY;
  const _Float16* Kb = Km + (size_t)(b * SEQ) * DMODEL + h * DKEY;
  const _Float16* Vb = V + (size_t)(b * SEQ) * DMODEL + h * DKEY;
  int m0 = wave * 16;

  // scores = Q K^T / 8 : wave owns one 16-row stripe, 8 column tiles
  {
    v8f acc[8] = {};
#pragma unroll
    for (int kk = 0; kk < DKEY; kk += 32) {
      v16h af = load_frag(Qb + (size_t)m0 * DMODEL + kk, DMODEL);
#pragma unroll
      for (int nt = 0; nt < 8; ++nt) {
        v16h bf = load_frag(Kb + (size_t)(nt * 16) * DMODEL + kk, DMODEL);
        acc[nt] = wmma16(af, bf, acc[nt]);
      }
    }
#pragma unroll
    for (int nt = 0; nt < 8; ++nt)
#pragma unroll
      for (int r = 0; r < 8; ++r)
        s_p[(m0 + r + 8 * hi) * SEQ + nt * 16 + nn] =
            (_Float16)(acc[nt][r] * 0.125f);
  }
  __syncthreads();

  if (threadIdx.x < SEQ) {           // row-wise softmax in f32
    int r = threadIdx.x;
    float mx = -1e30f;
    for (int c = 0; c < SEQ; ++c) {
      float v = (float)s_p[r * SEQ + c];
      mx = v > mx ? v : mx;
    }
    float sum = 0.f;
    for (int c = 0; c < SEQ; ++c) sum += __expf((float)s_p[r * SEQ + c] - mx);
    float inv = 1.f / sum;
    for (int c = 0; c < SEQ; ++c)
      s_p[r * SEQ + c] = (_Float16)(__expf((float)s_p[r * SEQ + c] - mx) * inv);
  }
  __syncthreads();

  // ctx = P(128x128) @ V(128x64): 4 column tiles share each A fragment
  {
    v8f acc[4] = {};
#pragma unroll
    for (int kk = 0; kk < SEQ; kk += 32) {
      v16h af = load_frag(s_p + m0 * SEQ + kk, SEQ);
#pragma unroll
      for (int nt = 0; nt < 4; ++nt) {
        v16h bf = load_frag_kn(Vb + (size_t)kk * DMODEL + nt * 16, DMODEL);
        acc[nt] = wmma16(af, bf, acc[nt]);
      }
    }
#pragma unroll
    for (int nt = 0; nt < 4; ++nt)
#pragma unroll
      for (int r = 0; r < 8; ++r) {
        int gn = b * SEQ + m0 + r + 8 * hi;
        ctx_h[(size_t)gn * DMODEL + h * DKEY + nt * 16 + nn] =
            (_Float16)acc[nt][r];
      }
  }
}

// ---------------------------------------------------------------------------
__global__ void ln_kernel(const float* __restrict__ y,
                          const float* __restrict__ g,
                          const float* __restrict__ be,
                          _Float16* __restrict__ outh) {
  __shared__ float red[DMODEL];
  int t = threadIdx.x;                        // DMODEL == blockDim.x == 256
  const float* yr = y + (size_t)blockIdx.x * DMODEL;
  float v = yr[t];
  red[t] = v; __syncthreads();
  for (int s2 = 128; s2 > 0; s2 >>= 1) {
    if (t < s2) red[t] += red[t + s2];
    __syncthreads();
  }
  float mu = red[0] * (1.f / DMODEL);
  __syncthreads();
  float d = v - mu;
  red[t] = d * d; __syncthreads();
  for (int s2 = 128; s2 > 0; s2 >>= 1) {
    if (t < s2) red[t] += red[t + s2];
    __syncthreads();
  }
  float var = red[0] * (1.f / DMODEL);
  float o = d * rsqrtf(var + 1e-5f) * g[t] + be[t];
  outh[(size_t)blockIdx.x * DMODEL + t] = (_Float16)o;
}

__global__ void head_kernel(const float* __restrict__ h1,
                            const float* __restrict__ w2,
                            const float* __restrict__ b2,
                            const int* __restrict__ tgt,
                            float* __restrict__ out) {
  int idx = blockIdx.x * 256 + threadIdx.x;   // (k, n, o) flat: 6*1024*2
  if (idx >= 6 * NBATCH * 2) return;
  int o = idx & 1;
  int n = (idx >> 1) & (NBATCH - 1);
  int k = idx / (2 * NBATCH);
  const float* hv = h1 + ((size_t)k * NBATCH + n) * HID;
  const float* wv = w2 + (size_t)(k * 2 + o) * HID;
  float s = b2[k * 2 + o];
  for (int i = 0; i < HID; ++i) s += hv[i] * wv[i];
  const int codes[6] = {99, 103, 105, 111, 115, 117};
  out[idx] = (tgt[n] == codes[k]) ? s : 0.f;
}

// ---------------------------------------------------------------------------
extern "C" void kernel_launch(void* const* d_in, const int* in_sizes, int n_in,
                              void* d_out, int out_size, void* d_ws,
                              size_t ws_size, hipStream_t stream) {
  (void)in_sizes; (void)n_in; (void)out_size; (void)ws_size;
  const int*   ctx = (const int*)d_in[0];
  const int*   tgt = (const int*)d_in[1];
  const float* emb = (const float*)d_in[2];
  // lstm_params pytree: dict keys sorted -> (bwd, fwd) per layer, tuples kept.
  const float *wih0b = (const float*)d_in[3],  *whh0b = (const float*)d_in[4],
              *bih0b = (const float*)d_in[5],  *bhh0b = (const float*)d_in[6];
  const float *wih0f = (const float*)d_in[7],  *whh0f = (const float*)d_in[8],
              *bih0f = (const float*)d_in[9],  *bhh0f = (const float*)d_in[10];
  const float *wih1b = (const float*)d_in[11], *whh1b = (const float*)d_in[12],
              *bih1b = (const float*)d_in[13], *bhh1b = (const float*)d_in[14];
  const float *wih1f = (const float*)d_in[15], *whh1f = (const float*)d_in[16],
              *bih1f = (const float*)d_in[17], *bhh1f = (const float*)d_in[18];
  const float *wq = (const float*)d_in[19], *bq = (const float*)d_in[20];
  const float *wk = (const float*)d_in[21], *bk = (const float*)d_in[22];
  const float *wv = (const float*)d_in[23], *bv = (const float*)d_in[24];
  const float *wo = (const float*)d_in[25], *bo = (const float*)d_in[26];
  const float *lng = (const float*)d_in[27], *lnb = (const float*)d_in[28];
  const float *w1 = (const float*)d_in[29], *b1 = (const float*)d_in[30];
  const float *w2 = (const float*)d_in[31], *b2 = (const float*)d_in[32];

  char* ws = (char*)d_ws;
  size_t off = 0;
  auto alloc = [&](size_t bytes) -> void* {
    void* p = ws + off;
    off += (bytes + 255) & ~(size_t)255;
    return p;
  };
  _Float16* x0_h   = (_Float16*)alloc((size_t)T_STEPS * NBATCH * HID * 2);
  _Float16* x1_h   = (_Float16*)alloc((size_t)T_STEPS * NBATCH * DMODEL * 2);
  _Float16* wih0f_h = (_Float16*)alloc(GATES * HID * 2);
  _Float16* whh0f_h = (_Float16*)alloc(GATES * HID * 2);
  _Float16* wih0b_h = (_Float16*)alloc(GATES * HID * 2);
  _Float16* whh0b_h = (_Float16*)alloc(GATES * HID * 2);
  _Float16* wih1f_h = (_Float16*)alloc(GATES * DMODEL * 2);
  _Float16* whh1f_h = (_Float16*)alloc(GATES * HID * 2);
  _Float16* wih1b_h = (_Float16*)alloc(GATES * DMODEL * 2);
  _Float16* whh1b_h = (_Float16*)alloc(GATES * HID * 2);
  _Float16* wq_h = (_Float16*)alloc(DMODEL * DMODEL * 2);
  _Float16* wk_h = (_Float16*)alloc(DMODEL * DMODEL * 2);
  _Float16* wv_h = (_Float16*)alloc(DMODEL * DMODEL * 2);
  _Float16* wo_h = (_Float16*)alloc(DMODEL * DMODEL * 2);
  _Float16* w1_h = (_Float16*)alloc(6 * HID * DMODEL * 2);
  float*    feats_f = (float*)alloc((size_t)NBATCH * DMODEL * 4);
  _Float16* feats_h = (_Float16*)alloc((size_t)NBATCH * DMODEL * 2);
  _Float16* q_h  = (_Float16*)alloc((size_t)NBATCH * DMODEL * 2);
  _Float16* k_h  = (_Float16*)alloc((size_t)NBATCH * DMODEL * 2);
  _Float16* v_h  = (_Float16*)alloc((size_t)NBATCH * DMODEL * 2);
  _Float16* ctx_h = (_Float16*)alloc((size_t)NBATCH * DMODEL * 2);
  float*    y_f  = (float*)alloc((size_t)NBATCH * DMODEL * 4);
  _Float16* ln_h = (_Float16*)alloc((size_t)NBATCH * DMODEL * 2);
  float*    h1_f = (float*)alloc((size_t)6 * NBATCH * HID * 4);

  auto cvt = [&](const float* s, _Float16* d, int n) {
    cvt_f16_kernel<<<(n + 255) / 256, 256, 0, stream>>>(s, d, n);
  };
  cvt(wih0f, wih0f_h, GATES * HID);   cvt(whh0f, whh0f_h, GATES * HID);
  cvt(wih0b, wih0b_h, GATES * HID);   cvt(whh0b, whh0b_h, GATES * HID);
  cvt(wih1f, wih1f_h, GATES * DMODEL); cvt(whh1f, whh1f_h, GATES * HID);
  cvt(wih1b, wih1b_h, GATES * DMODEL); cvt(whh1b, whh1b_h, GATES * HID);
  cvt(wq, wq_h, DMODEL * DMODEL); cvt(wk, wk_h, DMODEL * DMODEL);
  cvt(wv, wv_h, DMODEL * DMODEL); cvt(wo, wo_h, DMODEL * DMODEL);
  cvt(w1, w1_h, 6 * HID * DMODEL);

  embed_kernel<<<T_STEPS * NBATCH, HID, 0, stream>>>(ctx, emb, x0_h);

  dim3 lgrid(NBATCH / 32, 2);
  lstm_kernel<<<lgrid, 256, 0, stream>>>(x0_h, HID,
      wih0f_h, whh0f_h, bih0f, bhh0f, wih0b_h, whh0b_h, bih0b, bhh0b,
      x1_h, nullptr);
  lstm_kernel<<<lgrid, 256, 0, stream>>>(x1_h, DMODEL,
      wih1f_h, whh1f_h, bih1f, bhh1f, wih1b_h, whh1b_h, bih1b, bhh1b,
      nullptr, feats_f);

  cvt(feats_f, feats_h, NBATCH * DMODEL);

  dim3 pgrid(NBATCH / 64, DMODEL / 64);
  gemm_kernel<<<pgrid, 256, 0, stream>>>(feats_h, DMODEL, wq_h, DMODEL, bq,
      nullptr, 0, nullptr, q_h, DMODEL, DMODEL, 0);
  gemm_kernel<<<pgrid, 256, 0, stream>>>(feats_h, DMODEL, wk_h, DMODEL, bk,
      nullptr, 0, nullptr, k_h, DMODEL, DMODEL, 0);
  gemm_kernel<<<pgrid, 256, 0, stream>>>(feats_h, DMODEL, wv_h, DMODEL, bv,
      nullptr, 0, nullptr, v_h, DMODEL, DMODEL, 0);

  attn_kernel<<<8 * 4, 256, 0, stream>>>(q_h, k_h, v_h, ctx_h);

  gemm_kernel<<<pgrid, 256, 0, stream>>>(ctx_h, DMODEL, wo_h, DMODEL, bo,
      feats_f, DMODEL, y_f, nullptr, DMODEL, DMODEL, 0);

  ln_kernel<<<NBATCH, DMODEL, 0, stream>>>(y_f, lng, lnb, ln_h);

  dim3 cgrid(NBATCH / 64, HID / 64);
  for (int k = 0; k < 6; ++k)
    gemm_kernel<<<cgrid, 256, 0, stream>>>(ln_h, DMODEL,
        w1_h + (size_t)k * HID * DMODEL, DMODEL, b1 + k * HID,
        nullptr, 0, h1_f + (size_t)k * NBATCH * HID, nullptr, HID, DMODEL, 1);

  head_kernel<<<(6 * NBATCH * 2 + 255) / 256, 256, 0, stream>>>(
      h1_f, w2, b2, tgt, (float*)d_out);
}
